// DCGRUCell_22127671509775
// MI455X (gfx1250) — matrix operand order repaired and checked by reading, not compiled, not run on hardware
//
#include <hip/hip_runtime.h>

#define N_NODES 2048
#define BATCH   64
#define UNITS   64
#define F_IN    66                 // D_IN + UNITS
#define FB      (F_IN * BATCH)     // 4224
#define MROWS   (BATCH * N_NODES)  // 131072 GEMM rows
#define KREAL   330                // F_IN * M
#define KPAD    352                // padded to 11 * 32
#define KTILES  11
#define E_NNZ   65536

typedef __attribute__((ext_vector_type(16))) __bf16 bf16x16;
typedef __attribute__((ext_vector_type(8)))  float  v8f;

union ABu { bf16x16 v; uint4 q[2]; };

__device__ __forceinline__ unsigned short f2bf(float f) {
  union { float f; unsigned u; } a; a.f = f;
  unsigned u = a.u;
  u += 0x7FFFu + ((u >> 16) & 1u);   // round-to-nearest-even
  return (unsigned short)(u >> 16);
}

// ---------------- elementwise helpers ----------------

__global__ void k_fill_zero(float* __restrict__ p, long n) {
  long i = (long)blockIdx.x * blockDim.x + threadIdx.x;
  if (i < n) p[i] = 0.f;
}

__global__ void k_copy_neg(float* __restrict__ dst, const float* __restrict__ src, long n) {
  long i = (long)blockIdx.x * blockDim.x + threadIdx.x;
  if (i < n) dst[i] = -src[i];
}

__global__ void k_zero_pad(unsigned short* __restrict__ xcat) {
  long i = (long)blockIdx.x * blockDim.x + threadIdx.x;
  if (i >= (long)MROWS * (KPAD - KREAL)) return;
  long row = i / (KPAD - KREAL);
  int  j   = (int)(i - row * (KPAD - KREAL));
  xcat[row * KPAD + KREAL + j] = 0;
}

// x0[n][f][b] = concat(inputs, state [* mul]) ; layout [N, F, B], b contiguous
__global__ void k_build_x0(const float* __restrict__ inputs,
                           const float* __restrict__ st,
                           const float* __restrict__ mul,
                           float* __restrict__ x0) {
  long i = (long)blockIdx.x * blockDim.x + threadIdx.x;
  if (i >= (long)N_NODES * FB) return;
  int  b = (int)(i & 63);
  long t = i >> 6;
  int  f = (int)(t % F_IN);
  int  n = (int)(t / F_IN);
  float v;
  if (f < 2) {
    v = inputs[(long)b * (N_NODES * 2) + (long)n * 2 + f];
  } else {
    long si = (long)b * (N_NODES * UNITS) + (long)n * UNITS + (f - 2);
    v = st[si];
    if (mul) v *= mul[si];
  }
  x0[i] = v;
}

// ---------------- sparse diffusion: y[rows[e]] += alpha*vals[e]*x[cols[e]] ----------------
// One block per edge; 4224 f32 per edge; accumulation buffer (34.6MB) lives in L2.
__global__ void k_spmm(const int* __restrict__ rows, const int* __restrict__ cols,
                       const float* __restrict__ vals,
                       const float* __restrict__ x, float* __restrict__ y, float alpha) {
  int e = blockIdx.x;
  int r = rows[e], c = cols[e];
  float v = alpha * vals[e];
  const float4* xs = (const float4*)(x + (long)c * FB);
  float* yb = y + (long)r * FB;
  for (int i = threadIdx.x; i < FB / 4; i += blockDim.x) {
    float4 t = xs[i];
    atomicAdd(yb + i * 4 + 0, v * t.x);
    atomicAdd(yb + i * 4 + 1, v * t.y);
    atomicAdd(yb + i * 4 + 2, v * t.z);
    atomicAdd(yb + i * 4 + 3, v * t.w);
  }
}

// scatter x_m[n][f][b] -> xcat[(b*N+n)*KPAD + m*F_IN + f]  (bf16), LDS transpose
__global__ void k_scatter(const float* __restrict__ xm, unsigned short* __restrict__ xcat, int m) {
  __shared__ float tile[F_IN * 65];   // pad 64->65 to kill bank conflicts
  int n = blockIdx.x;
  const float* src = xm + (long)n * FB;
  for (int i = threadIdx.x; i < FB; i += blockDim.x) {
    int f = i >> 6, b = i & 63;
    tile[f * 65 + b] = src[i];
  }
  __syncthreads();
  for (int i = threadIdx.x; i < BATCH * F_IN; i += blockDim.x) {
    int b = i / F_IN, f = i - b * F_IN;
    long row = (long)b * N_NODES + n;
    xcat[row * KPAD + m * F_IN + f] = f2bf(tile[f * 65 + b]);
  }
}

// ---------------- W pre-swizzle into WMMA B-fragment layout ----------------
// frag = kt*ntiles + nt, 1KB each; lane L holds col n=L%16, halves K = 8*(L/16)+h (h<8), 16+8*(L/16)+h-8 (h>=8).
// k' = m*66+f remap  <->  original W row = f*5+m.
__global__ void k_pack_w(const float* __restrict__ W, unsigned short* __restrict__ dst,
                         int ncols, int ntiles) {
  int t = blockIdx.x * blockDim.x + threadIdx.x;
  if (t >= KTILES * ntiles * 32) return;
  int lane = t & 31, frag = t >> 5;
  int nt = frag % ntiles, kt = frag / ntiles;
  int n  = nt * 16 + (lane & 15);
  int hi = lane >> 4;
  unsigned short* o = dst + (long)frag * 512 + lane * 16;
  for (int h = 0; h < 16; ++h) {
    int kp = kt * 32 + (h < 8 ? hi * 8 + h : 16 + hi * 8 + (h - 8));
    float v = 0.f;
    if (kp < KREAL) {
      int m = kp / F_IN, f = kp - m * F_IN;
      v = W[(long)(f * 5 + m) * ncols + n];
    }
    o[h] = f2bf(v);
  }
}

// ---------------- bf16 WMMA GEMM + fused activation ----------------
// Each wave: 16 rows x 64 cols (4 x 16x16 tiles), K = 11 steps of 32.
// mode 0: sigmoid -> r (cols<64) / u (cols>=64).  mode 1: tanh -> out0.
__global__ void __launch_bounds__(256)
k_gemm(const unsigned short* __restrict__ A, const unsigned short* __restrict__ Bp,
       const float* __restrict__ bias, float* __restrict__ out0, float* __restrict__ out1,
       int ntiles, int mode) {
  int lane = threadIdx.x & 31, wave = threadIdx.x >> 5;
  int rowTile = blockIdx.x * 8 + wave;
  int ntBase  = blockIdx.y * 4;
  int m16 = lane & 15, hi = lane >> 4;
  long rowBase = (long)rowTile * 16;
  // A operand: lane holds row m16; halves = K {kb..kb+7, 16+kb..16+kb+7}, kb = 8*hi
  const unsigned short* aRow = A + (rowBase + m16) * KPAD + hi * 8;

  v8f acc[4] = {v8f{}, v8f{}, v8f{}, v8f{}};

  for (int kt = 0; kt < KTILES; ++kt) {
    ABu a;
    const uint4* ap = (const uint4*)(aRow + kt * 32);
    a.q[0] = ap[0];        // K kb..kb+7
    a.q[1] = ap[2];        // K 16+kb..16+kb+7
#pragma unroll
    for (int j = 0; j < 4; ++j) {
      ABu b;
      const uint4* bp = (const uint4*)(Bp + ((long)(kt * ntiles + ntBase + j)) * 512 + lane * 16);
      b.q[0] = bp[0];
      b.q[1] = bp[1];
      acc[j] = __builtin_amdgcn_wmma_f32_16x16x32_bf16(
          false, a.v, false, b.v, (short)0, acc[j], false, false);
    }
  }

#pragma unroll
  for (int j = 0; j < 4; ++j) {
    int col = (ntBase + j) * 16 + m16;
    float bsv = bias[col];
#pragma unroll
    for (int r = 0; r < 8; ++r) {
      long row = rowBase + r + hi * 8;   // C/D layout: VGPR r -> M=r (lanes 0-15), M=r+8 (16-31)
      float vv = acc[j][r] + bsv;
      if (mode == 0) {
        vv = 1.f / (1.f + __expf(-vv));
        if (col < UNITS) out0[row * UNITS + col] = vv;
        else             out1[row * UNITS + (col - UNITS)] = vv;
      } else {
        out0[row * UNITS + col] = tanhf(vv);
      }
    }
  }
}

// new_state = u*state + (1-u)*c
__global__ void k_final(const float* __restrict__ st, const float* __restrict__ u,
                        const float* __restrict__ c, float* __restrict__ out, long n) {
  long i = (long)blockIdx.x * blockDim.x + threadIdx.x;
  if (i < n) {
    float uu = u[i];
    out[i] = uu * st[i] + (1.f - uu) * c[i];
  }
}

// ---------------- host orchestration ----------------

static void run_phase(const float* inputs, const float* st, const float* mul,
                      const int* e1, const float* v1, const int* e2, const float* v2,
                      float* x0, float* xa, float* xb, unsigned short* xcat,
                      hipStream_t stream) {
  const long XN = (long)N_NODES * FB;
  unsigned gX = (unsigned)((XN + 255) / 256);

  k_build_x0<<<gX, 256, 0, stream>>>(inputs, st, mul, x0);
  k_scatter<<<N_NODES, 256, 0, stream>>>(x0, xcat, 0);

  // support 1: x1 = S1 x0 ; x2 = 2 S1 x1 - x0
  k_fill_zero<<<gX, 256, 0, stream>>>(xa, XN);
  k_spmm<<<E_NNZ, 256, 0, stream>>>(e1, e1 + E_NNZ, v1, x0, xa, 1.f);
  k_scatter<<<N_NODES, 256, 0, stream>>>(xa, xcat, 1);
  k_copy_neg<<<gX, 256, 0, stream>>>(xb, x0, XN);
  k_spmm<<<E_NNZ, 256, 0, stream>>>(e1, e1 + E_NNZ, v1, xa, xb, 2.f);
  k_scatter<<<N_NODES, 256, 0, stream>>>(xb, xcat, 2);

  // support 2: x3 = S2 x0 ; x4 = 2 S2 x3 - x0
  k_fill_zero<<<gX, 256, 0, stream>>>(xa, XN);
  k_spmm<<<E_NNZ, 256, 0, stream>>>(e2, e2 + E_NNZ, v2, x0, xa, 1.f);
  k_scatter<<<N_NODES, 256, 0, stream>>>(xa, xcat, 3);
  k_copy_neg<<<gX, 256, 0, stream>>>(xb, x0, XN);
  k_spmm<<<E_NNZ, 256, 0, stream>>>(e2, e2 + E_NNZ, v2, xa, xb, 2.f);
  k_scatter<<<N_NODES, 256, 0, stream>>>(xb, xcat, 4);
}

extern "C" void kernel_launch(void* const* d_in, const int* in_sizes, int n_in,
                              void* d_out, int out_size, void* d_ws, size_t ws_size,
                              hipStream_t stream) {
  const float* inputs = (const float*)d_in[0];
  const float* state  = (const float*)d_in[1];
  const int*   e1     = (const int*)d_in[2];
  const float* v1     = (const float*)d_in[3];
  const int*   e2     = (const int*)d_in[4];
  const float* v2     = (const float*)d_in[5];
  const float* Wg     = (const float*)d_in[6];
  const float* bg     = (const float*)d_in[7];
  const float* Wc     = (const float*)d_in[8];
  const float* bc     = (const float*)d_in[9];
  float* out = (float*)d_out;

  char* ws = (char*)d_ws;
  const size_t XB    = (size_t)N_NODES * FB * sizeof(float);            // 34.6 MB
  const size_t XCATB = (size_t)MROWS * KPAD * sizeof(unsigned short);   // 92.3 MB
  const size_t OB    = (size_t)MROWS * UNITS * sizeof(float);           // 33.55 MB

  float*          x0   = (float*)(ws);
  float*          xa   = (float*)(ws + XB);
  float*          xb   = (float*)(ws + 2 * XB);
  unsigned short* xcat = (unsigned short*)(ws + 3 * XB);
  float*          rb   = (float*)(ws + 3 * XB + XCATB);
  float*          ub   = (float*)(ws + 3 * XB + XCATB + OB);
  float*          cb   = (float*)(ws + 3 * XB + XCATB + 2 * OB);
  unsigned short* wgp  = (unsigned short*)(ws + 3 * XB + XCATB + 3 * OB);
  unsigned short* wcp  = wgp + (size_t)KTILES * 8 * 512;

  // one-time per launch: weight swizzle + K-pad zeros (scatters never touch pad)
  k_pack_w<<<(KTILES * 8 * 32 + 255) / 256, 256, 0, stream>>>(Wg, wgp, 2 * UNITS, 8);
  k_pack_w<<<(KTILES * 4 * 32 + 255) / 256, 256, 0, stream>>>(Wc, wcp, UNITS, 4);
  {
    long npad = (long)MROWS * (KPAD - KREAL);
    k_zero_pad<<<(unsigned)((npad + 255) / 256), 256, 0, stream>>>(xcat);
  }

  // phase 1: gate = sigmoid(dconv(inputs, state) @ Wg + bg) -> r, u
  run_phase(inputs, state, nullptr, e1, v1, e2, v2, x0, xa, xb, xcat, stream);
  k_gemm<<<dim3(MROWS / 16 / 8, 2), 256, 0, stream>>>(xcat, wgp, bg, rb, ub, 8, 0);

  // phase 2: c = tanh(dconv(inputs, r*state) @ Wc + bc)
  run_phase(inputs, state, rb, e1, v1, e2, v2, x0, xa, xb, xcat, stream);
  k_gemm<<<dim3(MROWS / 16 / 8, 1), 256, 0, stream>>>(xcat, wcp, bc, cb, nullptr, 4, 1);

  // new_state = u*state + (1-u)*c
  {
    long n = (long)MROWS * UNITS;
    k_final<<<(unsigned)((n + 255) / 256), 256, 0, stream>>>(state, ub, cb, out, n);
  }
}
